// DownsampleDISCO_32366873543343
// MI455X (gfx1250) — matrix operand order repaired and checked by reading, not compiled, not run on hardware
//
#include <hip/hip_runtime.h>

// DISCO downsample conv: x(4,96,512,512) f32, stride2 pad1 3x3, w = coeff . basis
// Strategy: pack effective weights as bf16 WMMA A-fragments (L2-resident, 162KB),
// stage 5-row input patch as bf16 in LDS (channel-contiguous, 4-wide vectorized),
// implicit GEMM via v_wmma_f32_16x16x32_bf16. Block = 12 waves, 96 och x 32 col x 2 row.

typedef __attribute__((ext_vector_type(16))) __bf16 v16bf;
typedef __attribute__((ext_vector_type(4)))  __bf16 v4bf;
typedef __attribute__((ext_vector_type(8)))  float  v8f;

#define CIN   96
#define COUT  96
#define NB    8      // basis count
#define HIN   512
#define WIN   512
#define HOUT  256
#define WOUT  256
#define OCH   6      // 96/16 out-channel chunks
#define KBLK  3      // 96/32 K blocks per tap
#define TAPS  9
#define NFRAG (TAPS*OCH*KBLK)            // 162 fragments
#define WPRE_DWORDS (NFRAG*32*8)         // 41472 dwords = 162 KB
#define PCOLS 66                          // 65 used + 1 pad
#define PROWS 5                           // 2 output rows -> 5 input rows
#define CHQ   (CIN/4)                     // 24 channel quads

__device__ __forceinline__ unsigned short f2bf(float f) {
    unsigned u = __float_as_uint(f);
    unsigned r = (u + 0x7FFFu + ((u >> 16) & 1u)) >> 16;   // RNE to bf16
    return (unsigned short)r;
}

// ---------------------------------------------------------------------------
// Kernel 1: w_eff[o,i,tap] = sum_b coeff[o,i,b]*basis[b,tap], emitted directly
// as 16x16x32 bf16 A-fragments in contiguous k order (k=16h+s, same order as the B reads):
//   (m,k): lane = m + 16*(k/16); vgpr j -> k=2j+16*hb
// ---------------------------------------------------------------------------
__global__ void disco_weight_pack(const float* __restrict__ coeff,
                                  const float* __restrict__ basis,
                                  unsigned* __restrict__ wpre) {
    int d = blockIdx.x * blockDim.x + threadIdx.x;  // 0..41471
    if (d >= WPRE_DWORDS) return;
    int j    = d & 7;
    int lane = (d >> 3) & 31;
    int frag = d >> 8;                 // (t*6+oc)*3+kb
    int kb   = frag % KBLK;
    int rest = frag / KBLK;
    int oc   = rest % OCH;
    int t    = rest / OCH;             // tap 0..8

    int m  = lane & 15;
    int hb = lane >> 4;
    int ke = 2*j + 16*hb;
    int o  = oc*16 + m;
    int i0 = kb*32 + ke;

    float w0 = 0.f, w1 = 0.f;
    #pragma unroll
    for (int b = 0; b < NB; ++b) {
        float bs = basis[b*TAPS + t];
        w0 += coeff[((size_t)o*CIN + i0    )*NB + b] * bs;
        w1 += coeff[((size_t)o*CIN + i0 + 1)*NB + b] * bs;
    }
    wpre[d] = (unsigned)f2bf(w0) | ((unsigned)f2bf(w1) << 16);
}

// ---------------------------------------------------------------------------
// Kernel 2: implicit-GEMM conv. Grid (8 wtiles, 128 row-pairs, 4 batch),
// 384 threads = 12 waves. Wave w: oc = w%6, wsub = w/6 (16-col subtile).
// LDS patch [5 rows][66 cols][96 ch] bf16; B-fragment = 32 contiguous bytes/lane.
// Each A fragment feeds two WMMAs (output rows h0 and h0+1).
// ---------------------------------------------------------------------------
__global__ void __launch_bounds__(384)
disco_conv_wmma(const float* __restrict__ x,
                const unsigned* __restrict__ wpre,
                const float* __restrict__ bias,
                float* __restrict__ out) {
    __shared__ __align__(32) __bf16 sB[PROWS * PCOLS * CIN];  // 63360 B

    const int tid = threadIdx.x;
    const int nb  = blockIdx.z;
    const int h0  = blockIdx.y * 2;         // first of 2 output rows
    const int wb0 = blockIdx.x * 32;        // first output col
    const size_t xbase = (size_t)nb * CIN * HIN * WIN;
    const int iw0 = 2*wb0 - 1;
    const size_t CHS = (size_t)HIN * WIN;   // channel stride in x

    // ---- stage 5 x 96 x 65 f32 -> bf16 LDS; 4 channels per work-item ----
    #pragma unroll
    for (int r = 0; r < PROWS; ++r) {
        const int ih  = 2*h0 - 1 + r;       // <= 511 always; < 0 only at top edge
        const int ihc = ih < 0 ? 0 : ih;
        // main cols 0..63: 24 chq x 64 cols = 1536 items -> 4 iterations
        for (int e = tid; e < CHQ * 64; e += 384) {
            int ch = (e >> 6) << 2;         // 4*chq
            int c  = e & 63;
            int iw = iw0 + c;               // < 512 always; < 0 only at left edge
            int iwc = iw < 0 ? 0 : iw;
            const float* xp = x + xbase + ((size_t)ch*HIN + ihc)*WIN + iwc;
            float v0 = xp[0];
            float v1 = xp[CHS];
            float v2 = xp[2*CHS];
            float v3 = xp[3*CHS];
            if (ih < 0 || iw < 0) { v0 = v1 = v2 = v3 = 0.f; }
            v4bf q = { (__bf16)v0, (__bf16)v1, (__bf16)v2, (__bf16)v3 };
            *(v4bf*)&sB[(r*PCOLS + c)*CIN + ch] = q;        // ds_store_b64
        }
    }
    // tail: column 64 (5 rows x 24 chq = 120 items)
    if (tid < PROWS * CHQ) {
        int r  = tid / CHQ;
        int ch = (tid - r*CHQ) << 2;
        int ih = 2*h0 - 1 + r;
        int ihc = ih < 0 ? 0 : ih;
        const float* xp = x + xbase + ((size_t)ch*HIN + ihc)*WIN + (2*wb0 + 63);
        float v0 = xp[0], v1 = xp[CHS], v2 = xp[2*CHS], v3 = xp[3*CHS];
        if (ih < 0) { v0 = v1 = v2 = v3 = 0.f; }
        v4bf q = { (__bf16)v0, (__bf16)v1, (__bf16)v2, (__bf16)v3 };
        *(v4bf*)&sB[(r*PCOLS + 64)*CIN + ch] = q;
    }
    __syncthreads();

    // ---- per-wave: 16 out-ch x 16 cols x 2 rows ----
    const int lane = tid & 31;
    const int wave = tid >> 5;
    const int oc   = wave % OCH;
    const int wsub = wave / OCH;            // 0 or 1
    const int ncol = lane & 15;
    const int khi  = lane >> 4;
    const int cbase = wsub*32 + 2*ncol;

    const v16bf* __restrict__ wfrag = (const v16bf*)wpre;  // 32 v16bf per fragment

    v8f acc0 = {}, acc1 = {};
    #pragma unroll
    for (int kh = 0; kh < 3; ++kh) {
        #pragma unroll
        for (int kw = 0; kw < 3; ++kw) {
            const int t = kh*3 + kw;
            #pragma unroll
            for (int kb = 0; kb < KBLK; ++kb) {
                const int frag = (t*OCH + oc)*KBLK + kb;
                v16bf a = wfrag[frag*32 + lane];                 // 2x global_load_b128 (L2-hot)
                const int boff = (cbase + kw)*CIN + kb*32 + khi*16;
                v16bf b0 = *(const v16bf*)&sB[(kh    *PCOLS)*CIN + boff];  // row h0
                acc0 = __builtin_amdgcn_wmma_f32_16x16x32_bf16(
                        false, a, false, b0, (short)0, acc0, false, false);
                v16bf b1 = *(const v16bf*)&sB[((kh+2)*PCOLS)*CIN + boff];  // row h0+1
                acc1 = __builtin_amdgcn_wmma_f32_16x16x32_bf16(
                        false, a, false, b1, (short)0, acc1, false, false);
            }
        }
    }

    // ---- store + bias (vectorized bias read, 32B aligned) ----
    const int  wcol = wb0 + wsub*16 + ncol;
    const int  obase = oc*16 + khi*8;
    const v8f  bv = *(const v8f*)(bias + obase);
    #pragma unroll
    for (int v = 0; v < 8; ++v) {
        size_t oidx = (((size_t)nb*COUT + obase + v)*HOUT + h0)*WOUT + wcol;
        out[oidx]        = acc0[v] + bv[v];   // row h0
        out[oidx + WOUT] = acc1[v] + bv[v];   // row h0+1
    }
}

// ---------------------------------------------------------------------------
extern "C" void kernel_launch(void* const* d_in, const int* in_sizes, int n_in,
                              void* d_out, int out_size, void* d_ws, size_t ws_size,
                              hipStream_t stream) {
    const float* x     = (const float*)d_in[0];
    const float* coeff = (const float*)d_in[1];
    const float* basis = (const float*)d_in[2];
    const float* bias  = (const float*)d_in[3];
    float*       out   = (float*)d_out;
    unsigned*    wpre  = (unsigned*)d_ws;   // 162 KB packed bf16 weight fragments

    disco_weight_pack<<<dim3(WPRE_DWORDS / 256), dim3(256), 0, stream>>>(coeff, basis, wpre);

    dim3 grid(WOUT / 32, HOUT / 2, 4);
    disco_conv_wmma<<<grid, dim3(384), 0, stream>>>(x, wpre, bias, out);
}
